// S4Block_48146583388753
// MI455X (gfx1250) — matrix operand order repaired and checked by reading, not compiled, not run on hardware
//
#include <hip/hip_runtime.h>
#include <hip/hip_bf16.h>

// ---------------------------------------------------------------------------
// S4 block for MI455X (gfx1250, wave32, WMMA).
//   B=8, H=128, L=16384, N=16, COND=4
// Pipeline:
//   1) precompute: weights -> bf16, SSM coefficients (lambda, lambda^T, Ceff)
//   2) GEMM1 (WMMA bf16): u = prelu(lin_w @ x + lin_b)
//   3) 3-phase blocked scan replacing the FFT conv (diagonal SSM recurrence)
//   4) GEMM2 (WMMA bf16): z = out_w @ y + out_b   -> d_out
//   5) BN stats per channel, then FiLM + prelu + residual in place.
// ---------------------------------------------------------------------------

static constexpr int Bb   = 8;
static constexpr int Hc   = 128;
static constexpr int Lc   = 16384;
static constexpr int Nst  = 16;
static constexpr int Tch  = 256;           // chunk length for blocked scan
static constexpr int Cch  = Lc / Tch;      // 64 chunks

typedef __attribute__((ext_vector_type(16))) __bf16 v16bf;
typedef __attribute__((ext_vector_type(8)))  float  v8f;

// ---------------------------------------------------------------------------
// Kernel 1: precompute bf16 weights + SSM coefficients.
// coef layout (each 2048 = H*N floats):
//   [0]=lam_re [1]=lam_im [2]=lamT_re [3]=lamT_im [4]=ce_re [5]=ce_im
// ---------------------------------------------------------------------------
__global__ __launch_bounds__(256) void precompute_k(
    const float* __restrict__ lin_w, const float* __restrict__ out_w,
    const float* __restrict__ log_dt, const float* __restrict__ log_A_real,
    const float* __restrict__ A_imag, const float* __restrict__ C_re,
    const float* __restrict__ C_im,
    __bf16* __restrict__ w1, __bf16* __restrict__ w2,
    float* __restrict__ coef)
{
    int tid = blockIdx.x * blockDim.x + threadIdx.x;   // 16384 threads
    for (int i = tid; i < Hc * Hc; i += gridDim.x * blockDim.x) {
        w1[i] = (__bf16)lin_w[i];
        w2[i] = (__bf16)out_w[i];
    }
    if (tid < Hc * Nst) {
        int h = tid >> 4;
        float dt  = __expf(log_dt[h]);
        float ar  = -__expf(log_A_real[tid]);
        float ai  = A_imag[tid];
        float dre = dt * ar, dim = dt * ai;
        // lambda = exp(dtA)
        float er  = __expf(dre);
        float lre = er * __cosf(dim);
        float lim = er * __sinf(dim);
        // lambda^T = exp(dtA * T)
        float erT  = __expf(dre * (float)Tch);
        float ltre = erT * __cosf(dim * (float)Tch);
        float ltim = erT * __sinf(dim * (float)Tch);
        // dB = (exp(dtA) - 1) / A   (complex divide by A)
        float nr = lre - 1.0f, ni = lim;
        float inv = 1.0f / (ar * ar + ai * ai);
        float dbr = (nr * ar + ni * ai) * inv;
        float dbi = (ni * ar - nr * ai) * inv;
        // Ceff = (C_re + i C_im) * dB
        float cr = C_re[tid], ci = C_im[tid];
        float cer = cr * dbr - ci * dbi;
        float cei = cr * dbi + ci * dbr;
        coef[0 * 2048 + tid] = lre;
        coef[1 * 2048 + tid] = lim;
        coef[2 * 2048 + tid] = ltre;
        coef[3 * 2048 + tid] = ltim;
        coef[4 * 2048 + tid] = cer;
        coef[5 * 2048 + tid] = cei;
    }
}

// ---------------------------------------------------------------------------
// Kernel 2/6: 128x128 GEMM over L columns, WMMA bf16, f32 accumulate.
//   out[b,g,l] = act( sum_h w[g,h] * in[b,h,l] + bias[g] )
// One block = one batch b and one 128-wide strip of l.
// 8 waves; wave w owns 16 l-columns, loops over 8 g-tiles, K=128 in 4 steps.
// Fragment packing per CDNA5 ISA 7.12.2 (A 16x32 bf16, B 32x16 bf16, C/D f32).
// ---------------------------------------------------------------------------
__global__ __launch_bounds__(256) void gemm_wmma_k(
    const float* __restrict__ in, const __bf16* __restrict__ wbf,
    const float* __restrict__ bias, float* __restrict__ out,
    const float* __restrict__ prelu_a, int use_prelu)
{
    __shared__ __bf16 Xs[128][136];        // padded: 34816 B LDS

    int blk = blockIdx.x;
    int b   = blk / (Lc / 128);
    int l0  = (blk % (Lc / 128)) * 128;
    int t   = threadIdx.x;

    // Stage the 128(h) x 128(l) tile of input, converting f32 -> bf16.
    for (int rep = 0; rep < 64; ++rep) {
        int idx = rep * 256 + t;
        int h = idx >> 7;
        int l = idx & 127;
        Xs[h][l] = (__bf16)in[((size_t)b * Hc + h) * Lc + l0 + l];
    }
    __syncthreads();

    int lane = t & 31;
    int wave = t >> 5;
    int n0   = wave * 16;                  // this wave's l-columns
    int m    = lane & 15;                  // A-matrix row (output channel in tile)
    int kb   = (lane < 16) ? 0 : 8;        // A K-window per lane half
    int gofs = (lane < 16) ? 0 : 8;        // C/D M offset per lane half
    int n    = lane & 15;                  // C/D column
    float a_scal = use_prelu ? prelu_a[0] : 0.0f;

    for (int gt = 0; gt < 8; ++gt) {
        int g0 = gt * 16;
        v8f acc = {0.f, 0.f, 0.f, 0.f, 0.f, 0.f, 0.f, 0.f};
        #pragma unroll
        for (int ks = 0; ks < 128; ks += 32) {
            v16bf afrag, bfrag;
            const __bf16* wr = wbf + (size_t)(g0 + m) * Hc + ks + kb;
            #pragma unroll
            for (int i = 0; i < 8; ++i) {
                afrag[i]     = wr[i];        // K = kb..kb+7
                afrag[8 + i] = wr[16 + i];   // K = kb+16..kb+23
            }
            const __bf16* xr = &Xs[ks + kb][n0 + n];   // B column n, K = kb..kb+7, kb+16..kb+23
            #pragma unroll
            for (int i = 0; i < 8; ++i) {
                bfrag[i]     = xr[i * 136];
                bfrag[8 + i] = xr[(16 + i) * 136];
            }
            acc = __builtin_amdgcn_wmma_f32_16x16x32_bf16(
                false, afrag, false, bfrag, (short)0, acc, false, false);
        }
        #pragma unroll
        for (int r = 0; r < 8; ++r) {
            int g = g0 + r + gofs;
            float v = acc[r] + bias[g];
            if (use_prelu) v = (v > 0.0f) ? v : a_scal * v;
            out[((size_t)b * Hc + g) * Lc + l0 + n0 + n] = v;
        }
    }
}

// ---------------------------------------------------------------------------
// Kernel 3: blocked-scan phase A — per-chunk local state with s0 = 0.
// thread = (b,h,chunk); writes end-state c[chunk] (16 complex) to cst.
// ---------------------------------------------------------------------------
__global__ __launch_bounds__(256) void scanA_k(
    const float* __restrict__ u, const float* __restrict__ coef,
    float* __restrict__ cst)
{
    int tid   = blockIdx.x * blockDim.x + threadIdx.x;  // B*H*C = 65536
    int chunk = tid & (Cch - 1);
    int bh    = tid / Cch;
    int h     = bh & (Hc - 1);

    float lre[Nst], lim[Nst], sre[Nst], sim[Nst];
    #pragma unroll
    for (int nn = 0; nn < Nst; ++nn) {
        int c = h * Nst + nn;
        lre[nn] = coef[c];
        lim[nn] = coef[2048 + c];
        sre[nn] = 0.0f; sim[nn] = 0.0f;
    }
    const float* up = u + (size_t)bh * Lc + (size_t)chunk * Tch;
    for (int j = 0; j < Tch; ++j) {
        float uu = up[j];
        #pragma unroll
        for (int nn = 0; nn < Nst; ++nn) {
            float r  = fmaf(lre[nn], sre[nn], fmaf(-lim[nn], sim[nn], uu));
            float ii = fmaf(lre[nn], sim[nn], lim[nn] * sre[nn]);
            sre[nn] = r; sim[nn] = ii;
        }
    }
    float* cp = cst + (size_t)tid * 32;
    #pragma unroll
    for (int nn = 0; nn < Nst; ++nn) { cp[2 * nn] = sre[nn]; cp[2 * nn + 1] = sim[nn]; }
}

// ---------------------------------------------------------------------------
// Kernel 4: phase B — sequential prefix over the 64 chunks (in place):
//   S0[chunk] = state entering chunk;  s <- lambda^T * s + c[chunk]
// thread = (b,h,n); 16384 threads.
// ---------------------------------------------------------------------------
__global__ __launch_bounds__(256) void scanB_k(
    const float* __restrict__ coef, float* __restrict__ cst)
{
    int tid = blockIdx.x * blockDim.x + threadIdx.x;   // B*H*N = 16384
    int nn  = tid & (Nst - 1);
    int bh  = tid / Nst;
    int h   = bh & (Hc - 1);
    int c   = h * Nst + nn;
    float ltr = coef[2 * 2048 + c];
    float lti = coef[3 * 2048 + c];
    float sr = 0.0f, si = 0.0f;
    float* base = cst + (size_t)bh * Cch * 32 + 2 * nn;
    for (int ch = 0; ch < Cch; ++ch) {
        float cr = base[(size_t)ch * 32];
        float ci = base[(size_t)ch * 32 + 1];
        base[(size_t)ch * 32]     = sr;      // state entering this chunk
        base[(size_t)ch * 32 + 1] = si;
        float nr = fmaf(ltr, sr, fmaf(-lti, si, cr));
        si = fmaf(ltr, si, fmaf(lti, sr, ci));
        sr = nr;
    }
}

// ---------------------------------------------------------------------------
// Kernel 5: phase C — replay each chunk from its entering state, emit
//   y[l] = 2*Re(sum_n Ceff_n * s_n[l]) + D_h * u[l]   (in place over u)
// ---------------------------------------------------------------------------
__global__ __launch_bounds__(256) void scanC_k(
    float* __restrict__ u, const float* __restrict__ coef,
    const float* __restrict__ cst, const float* __restrict__ Dv)
{
    int tid   = blockIdx.x * blockDim.x + threadIdx.x;  // 65536
    int chunk = tid & (Cch - 1);
    int bh    = tid / Cch;
    int h     = bh & (Hc - 1);

    float lre[Nst], lim[Nst], cer[Nst], cei[Nst], sre[Nst], sim[Nst];
    const float* S0 = cst + (size_t)tid * 32;
    #pragma unroll
    for (int nn = 0; nn < Nst; ++nn) {
        int c = h * Nst + nn;
        lre[nn] = coef[c];
        lim[nn] = coef[2048 + c];
        cer[nn] = coef[4 * 2048 + c];
        cei[nn] = coef[5 * 2048 + c];
        sre[nn] = S0[2 * nn];
        sim[nn] = S0[2 * nn + 1];
    }
    float Dh = Dv[h];
    float* up = u + (size_t)bh * Lc + (size_t)chunk * Tch;
    for (int j = 0; j < Tch; ++j) {
        float uu  = up[j];
        float acc = 0.0f;
        #pragma unroll
        for (int nn = 0; nn < Nst; ++nn) {
            float r  = fmaf(lre[nn], sre[nn], fmaf(-lim[nn], sim[nn], uu));
            float ii = fmaf(lre[nn], sim[nn], lim[nn] * sre[nn]);
            sre[nn] = r; sim[nn] = ii;
            acc = fmaf(cer[nn], r, fmaf(-cei[nn], ii, acc));
        }
        up[j] = fmaf(2.0f, acc, Dh * uu);
    }
}

// ---------------------------------------------------------------------------
// Kernel 7: BatchNorm stats per channel g over (B, L).
// stats[g] = mean, stats[128+g] = rsqrt(var + eps)
// ---------------------------------------------------------------------------
__global__ __launch_bounds__(256) void bnstats_k(
    const float* __restrict__ z, float* __restrict__ stats)
{
    int g = blockIdx.x;
    int t = threadIdx.x;
    float s = 0.0f, s2 = 0.0f;
    for (int b = 0; b < Bb; ++b) {
        const float* p = z + ((size_t)b * Hc + g) * Lc;
        for (int l = t; l < Lc; l += 256) {
            float v = p[l];
            s += v; s2 = fmaf(v, v, s2);
        }
    }
    __shared__ float sh[256], sh2[256];
    sh[t] = s; sh2[t] = s2;
    __syncthreads();
    for (int ofs = 128; ofs > 0; ofs >>= 1) {
        if (t < ofs) { sh[t] += sh[t + ofs]; sh2[t] += sh2[t + ofs]; }
        __syncthreads();
    }
    if (t == 0) {
        float cnt = (float)Bb * (float)Lc;
        float m   = sh[0] / cnt;
        float var = sh2[0] / cnt - m * m;
        stats[g]        = m;
        stats[Hc + g]   = rsqrtf(var + 1e-5f);
    }
}

// ---------------------------------------------------------------------------
// Kernel 8: BN apply + FiLM (tiny 4-wide dot per element) + PReLU + residual.
// In place over d_out.
// ---------------------------------------------------------------------------
__global__ __launch_bounds__(256) void finalize_k(
    float* __restrict__ z, const float* __restrict__ x,
    const float* __restrict__ stats, const float* __restrict__ cond,
    const float* __restrict__ film_w, const float* __restrict__ film_b,
    const float* __restrict__ prelu2_a, const float* __restrict__ res_w)
{
    size_t idx = (size_t)blockIdx.x * 256 + threadIdx.x;   // one elem/thread
    int b = (int)(idx / ((size_t)Hc * Lc));
    int g = (int)((idx / Lc) & (Hc - 1));
    float a2 = prelu2_a[0];

    float gain = film_b[g];
    float bias = film_b[Hc + g];
    #pragma unroll
    for (int j = 0; j < 4; ++j) {
        float cj = cond[b * 4 + j];
        gain = fmaf(cj, film_w[g * 4 + j], gain);
        bias = fmaf(cj, film_w[(Hc + g) * 4 + j], bias);
    }
    float v = (z[idx] - stats[g]) * stats[Hc + g];
    v = fmaf(v, gain, bias);
    v = (v > 0.0f) ? v : a2 * v;
    z[idx] = fmaf(res_w[g], x[idx], v);
}

// ---------------------------------------------------------------------------
extern "C" void kernel_launch(void* const* d_in, const int* in_sizes, int n_in,
                              void* d_out, int out_size, void* d_ws, size_t ws_size,
                              hipStream_t stream)
{
    (void)in_sizes; (void)n_in; (void)out_size; (void)ws_size;

    const float* x      = (const float*)d_in[0];
    const float* cond   = (const float*)d_in[1];
    const float* lin_w  = (const float*)d_in[2];
    const float* lin_b  = (const float*)d_in[3];
    const float* prelu1 = (const float*)d_in[4];
    const float* log_dt = (const float*)d_in[5];
    const float* logAre = (const float*)d_in[6];
    const float* Aim    = (const float*)d_in[7];
    const float* C_re   = (const float*)d_in[8];
    const float* C_im   = (const float*)d_in[9];
    const float* Dv     = (const float*)d_in[10];
    const float* out_w  = (const float*)d_in[11];
    const float* out_b  = (const float*)d_in[12];
    const float* film_w = (const float*)d_in[13];
    const float* film_b = (const float*)d_in[14];
    const float* prelu2 = (const float*)d_in[15];
    const float* res_w  = (const float*)d_in[16];
    float* out = (float*)d_out;

    char* ws = (char*)d_ws;
    size_t off = 0;
    float*  u    = (float*)(ws + off); off += (size_t)Bb * Hc * Lc * 4;          // 64 MB
    float*  cst  = (float*)(ws + off); off += (size_t)Bb * Hc * Cch * Nst * 8;   // 8 MB
    float*  coef = (float*)(ws + off); off += 6 * 2048 * 4;                      // 48 KB
    __bf16* w1   = (__bf16*)(ws + off); off += (size_t)Hc * Hc * 2;              // 32 KB
    __bf16* w2   = (__bf16*)(ws + off); off += (size_t)Hc * Hc * 2;              // 32 KB
    float*  stats = (float*)(ws + off); off += 2 * Hc * 4;                       // 1 KB

    // 1) coefficients + bf16 weights
    precompute_k<<<64, 256, 0, stream>>>(lin_w, out_w, log_dt, logAre, Aim,
                                         C_re, C_im, w1, w2, coef);
    // 2) input linear + PReLU  (WMMA)
    gemm_wmma_k<<<Bb * (Lc / 128), 256, 0, stream>>>(x, w1, lin_b, u, prelu1, 1);
    // 3-5) blocked scan (replaces FFT convolution); y overwrites u in place
    scanA_k<<<(Bb * Hc * Cch) / 256, 256, 0, stream>>>(u, coef, cst);
    scanB_k<<<(Bb * Hc * Nst) / 256, 256, 0, stream>>>(coef, cst);
    scanC_k<<<(Bb * Hc * Cch) / 256, 256, 0, stream>>>(u, coef, cst, Dv);
    // 6) output linear (WMMA) -> d_out
    gemm_wmma_k<<<Bb * (Lc / 128), 256, 0, stream>>>(u, w2, out_b, out, prelu1, 0);
    // 7) batchnorm stats
    bnstats_k<<<Hc, 256, 0, stream>>>(out, stats);
    // 8) BN apply + FiLM + PReLU + residual, in place
    finalize_k<<<(Bb * Hc * Lc) / 256, 256, 0, stream>>>(out, x, stats, cond,
                                                         film_w, film_b, prelu2, res_w);
}
